// TreeEncoder_30623116820880
// MI455X (gfx1250) — matrix operand order repaired and checked by reading, not compiled, not run on hardware
//
#include <hip/hip_runtime.h>
#include <hip/hip_bf16.h>

// TreeLSTM (4-ary, complete tree) for MI455X / gfx1250.
// All GEMM work (x@Wioux, hsum@Wiouh, x@Wfx, ch_j@Wfh) runs on
// v_wmma_f32_16x16x32_bf16 (bf16 multiplicands, f32 accumulate).
// Levels are processed leaves->root as separate launches on `stream`,
// which provides the inter-level dependency.

typedef __attribute__((ext_vector_type(16))) __bf16 v16bf;
typedef __attribute__((ext_vector_type(8)))  __bf16 v8bf;
typedef __attribute__((ext_vector_type(8)))  float  v8f;
typedef __attribute__((ext_vector_type(4)))  float  f32x4;

#define LDS_ST 264   // LDS row stride (bf16 elems); 528B rows keep 16B alignment

__device__ __forceinline__ float sigmoidf_(float x) { return 1.0f / (1.0f + __expf(-x)); }

__device__ __forceinline__ v8f wmma_bf16(v16bf a, v16bf b, v8f c) {
  // 8 args: (neg_a, A, neg_b, B, c_mod, C, reuse_a, reuse_b)
  return __builtin_amdgcn_wmma_f32_16x16x32_bf16(false, a, false, b, (short)0, c,
                                                 false, false);
}

// A-matrix (bf16 16x32) fragment from row-major LDS tile [16][LDS_ST]:
//   lanes 0-15 : row M=lane,    elems = K[0..7],  K[16..23]
//   lanes16-31 : row M=lane-16, elems = K[8..15], K[24..31]
__device__ __forceinline__ v16bf load_a_frag(const __bf16* S, int k32, int lane) {
  int row = lane & 15;
  int kb  = (lane >> 4) << 3;
  const __bf16* p = S + row * LDS_ST + (k32 << 5) + kb;
  v8bf lo = *(const v8bf*)(p);
  v8bf hi = *(const v8bf*)(p + 16);
  return __builtin_shufflevector(lo, hi, 0,1,2,3,4,5,6,7,8,9,10,11,12,13,14,15);
}

// ---------------------------------------------------------------------------
// Pack fp32 weight W[K=256][C] into bf16 B-fragment order:
//   flat[((kt*Ctiles + ct)*32 + lane)*16 + i] =
//       bf16( W[kt*32 + (lane>>4)*8 + (i&7) + (i>>3)*16][ct*16 + (lane&15)] )
// so phase-2 B loads are a single contiguous 32B load per lane.
// ---------------------------------------------------------------------------
__global__ void pack_weight_kernel(const float* __restrict__ W,
                                   __bf16* __restrict__ out,
                                   int Ctiles, int C, int total) {
  int idx = blockIdx.x * blockDim.x + threadIdx.x;
  if (idx >= total) return;
  int i    = idx & 15;
  int lane = (idx >> 4) & 31;
  int ct   = (idx >> 9) % Ctiles;
  int kt   = idx / (Ctiles << 9);
  int k    = (kt << 5) + ((lane >> 4) << 3) + (i & 7) + ((i >> 3) << 4);
  int col  = (ct << 4) + (lane & 15);
  out[idx] = (__bf16)W[(size_t)k * C + col];
}

// ---------------------------------------------------------------------------
// One level of the tree. Block = 256 threads (8 waves) = 16 nodes.
// LEAF=true skips all child staging / child GEMMs (largest level).
// ---------------------------------------------------------------------------
template <bool LEAF>
__global__ __launch_bounds__(256)
void tree_level_kernel(const int* __restrict__ tokens,
                       const float* __restrict__ emb,
                       const v16bf* __restrict__ Wioux_p,  // [8][48][32] frags
                       const v16bf* __restrict__ Wiouh_p,  // [8][48][32]
                       const v16bf* __restrict__ Wfx_p,    // [8][16][32]
                       const v16bf* __restrict__ Wfh_p,    // [8][16][32]
                       const float* __restrict__ bioux,
                       const float* __restrict__ biouh,
                       const float* __restrict__ bfx,
                       const float* __restrict__ bfh,
                       float* __restrict__ h,              // d_out, [N][256]
                       float* __restrict__ c,              // ws,    [N][256]
                       int s, int sz, int N) {
  extern __shared__ char smem[];
  __bf16* Sx  = (__bf16*)smem;              // x rows (bf16)         [16][LDS_ST]
  __bf16* Shs = Sx + 16 * LDS_ST;           // sum of children h     [16][LDS_ST]
  __bf16* Sch = Shs + 16 * LDS_ST;          // children h, [4][16][LDS_ST]

  const int tile     = blockIdx.x;
  const int nodebase = s + tile * 16;
  const int t        = threadIdx.x;
  const int r        = t >> 4;              // staging row 0..15
  const int cb       = (t & 15) << 4;       // staging col segment (16 floats)
  const int node     = nodebase + r;
  const bool valid   = node < (s + sz);

  // ---- Phase 1: stage A-matrices into LDS ------------------------------
  {
    int tok = valid ? tokens[node] : 0;
    const float* er = emb + (size_t)tok * 256 + cb;
#pragma unroll
    for (int q = 0; q < 4; ++q) {
      f32x4 v = {0.f, 0.f, 0.f, 0.f};
      if (valid) v = *(const f32x4*)(er + q * 4);
#pragma unroll
      for (int e = 0; e < 4; ++e) Sx[r * LDS_ST + cb + q * 4 + e] = (__bf16)v[e];
    }
  }
  if (!LEAF) {
    float acc[16];
#pragma unroll
    for (int e = 0; e < 16; ++e) acc[e] = 0.f;
#pragma unroll
    for (int j = 0; j < 4; ++j) {
      int child = node * 4 + 1 + j;
      bool cv = valid && (child < N);
      const float* hr = h + (size_t)child * 256 + cb;
      __bf16* Sc = Sch + (j * 16 + r) * LDS_ST + cb;
#pragma unroll
      for (int q = 0; q < 4; ++q) {
        f32x4 v = {0.f, 0.f, 0.f, 0.f};
        if (cv) v = *(const f32x4*)(hr + q * 4);
#pragma unroll
        for (int e = 0; e < 4; ++e) {
          Sc[q * 4 + e] = (__bf16)v[e];
          acc[q * 4 + e] += v[e];
        }
      }
    }
#pragma unroll
    for (int e = 0; e < 16; ++e) Shs[r * LDS_ST + cb + e] = (__bf16)acc[e];
  }
  __syncthreads();

  // ---- Phase 2: WMMA GEMMs, uniform control flow (EXEC all-1s) ---------
  const int wave = t >> 5;
  const int lane = t & 31;

  for (int pass = 0; pass < 2; ++pass) {
    const int dt = wave + pass * 8;         // 16-col slab index, 0..15
    v8f ai = {}, ao = {}, au = {}, axf = {};
    v8f af[4] = {{}, {}, {}, {}};

#pragma unroll
    for (int k32 = 0; k32 < 8; ++k32) {
      v16bf a_x = load_a_frag(Sx, k32, lane);
      ai = wmma_bf16(a_x, Wioux_p[(k32 * 48 +      dt) * 32 + lane], ai);
      ao = wmma_bf16(a_x, Wioux_p[(k32 * 48 + 16 + dt) * 32 + lane], ao);
      au = wmma_bf16(a_x, Wioux_p[(k32 * 48 + 32 + dt) * 32 + lane], au);
      if (!LEAF) {
        v16bf a_h = load_a_frag(Shs, k32, lane);
        ai = wmma_bf16(a_h, Wiouh_p[(k32 * 48 +      dt) * 32 + lane], ai);
        ao = wmma_bf16(a_h, Wiouh_p[(k32 * 48 + 16 + dt) * 32 + lane], ao);
        au = wmma_bf16(a_h, Wiouh_p[(k32 * 48 + 32 + dt) * 32 + lane], au);
        axf = wmma_bf16(a_x, Wfx_p[(k32 * 16 + dt) * 32 + lane], axf);
#pragma unroll
        for (int j = 0; j < 4; ++j) {
          v16bf a_c = load_a_frag(Sch + j * 16 * LDS_ST, k32, lane);
          af[j] = wmma_bf16(a_c, Wfh_p[(k32 * 16 + dt) * 32 + lane], af[j]);
        }
      }
    }

    // ---- Epilogue: gates + cell update (C/D layout: M=v+8*(lane>>4)) ---
    const int nc   = lane & 15;
    const int hs   = lane >> 4;
    const int dcol = dt * 16 + nc;
    const float bi = bioux[dcol]       + biouh[dcol];
    const float bo = bioux[256 + dcol] + biouh[256 + dcol];
    const float bu = bioux[512 + dcol] + biouh[512 + dcol];
    const float bf = LEAF ? 0.f : (bfx[dcol] + bfh[dcol]);

#pragma unroll
    for (int v = 0; v < 8; ++v) {
      int M  = v + 8 * hs;
      int nd = nodebase + M;
      if (nd < s + sz) {
        float gi = sigmoidf_(ai[v] + bi);
        float go = sigmoidf_(ao[v] + bo);
        float gu = tanhf(au[v] + bu);
        float cn = gi * gu;
        if (!LEAF) {
#pragma unroll
          for (int j = 0; j < 4; ++j) {
            int child = nd * 4 + 1 + j;
            if (child < N) {
              float f = sigmoidf_(af[j][v] + axf[v] + bf);
              cn += f * c[(size_t)child * 256 + dcol];
            }
          }
        }
        c[(size_t)nd * 256 + dcol] = cn;
        h[(size_t)nd * 256 + dcol] = go * tanhf(cn);
      }
    }
  }
}

// ---------------------------------------------------------------------------
extern "C" void kernel_launch(void* const* d_in, const int* in_sizes, int n_in,
                              void* d_out, int out_size, void* d_ws, size_t ws_size,
                              hipStream_t stream) {
  const int*   tokens = (const int*)d_in[0];
  // d_in[1] = branching (device scalar); complete 4-ary tree per reference.
  const float* emb    = (const float*)d_in[2];
  const float* Wioux  = (const float*)d_in[3];
  const float* bioux  = (const float*)d_in[4];
  const float* Wiouh  = (const float*)d_in[5];
  const float* biouh  = (const float*)d_in[6];
  const float* Wfx    = (const float*)d_in[7];
  const float* bfx    = (const float*)d_in[8];
  const float* Wfh    = (const float*)d_in[9];
  const float* bfh    = (const float*)d_in[10];

  const int N = in_sizes[0];
  float* h = (float*)d_out;

  // Workspace: c buffer (N*256 f32) then packed bf16 weight fragments.
  char* ws = (char*)d_ws;
  float* cbuf = (float*)ws;
  size_t off = (size_t)N * 256 * sizeof(float);
  off = (off + 255) & ~(size_t)255;
  __bf16* Wioux_p = (__bf16*)(ws + off); off += (size_t)256 * 768 * 2;
  __bf16* Wiouh_p = (__bf16*)(ws + off); off += (size_t)256 * 768 * 2;
  __bf16* Wfx_p   = (__bf16*)(ws + off); off += (size_t)256 * 256 * 2;
  __bf16* Wfh_p   = (__bf16*)(ws + off); off += (size_t)256 * 256 * 2;

  // 1) pack weights -> bf16 fragment order (shared by every level kernel)
  {
    int tot_iou = 8 * 48 * 512;   // 196608 = 256*768
    int tot_f   = 8 * 16 * 512;   //  65536 = 256*256
    pack_weight_kernel<<<(tot_iou + 255) / 256, 256, 0, stream>>>(Wioux, Wioux_p, 48, 768, tot_iou);
    pack_weight_kernel<<<(tot_iou + 255) / 256, 256, 0, stream>>>(Wiouh, Wiouh_p, 48, 768, tot_iou);
    pack_weight_kernel<<<(tot_f   + 255) / 256, 256, 0, stream>>>(Wfx,   Wfx_p,   16, 256, tot_f);
    pack_weight_kernel<<<(tot_f   + 255) / 256, 256, 0, stream>>>(Wfh,   Wfh_p,   16, 256, tot_f);
  }

  // 2) BFS level boundaries of complete 4-ary tree, processed leaves->root.
  int starts[20], sizes[20], L = 0;
  {
    long st = 0, szl = 1;
    while (st < N && L < 20) {
      starts[L] = (int)st;
      sizes[L]  = (int)((st + szl <= N) ? szl : (N - st));
      ++L; st += szl; szl *= 4;
    }
  }

  const size_t smem_leaf  = (size_t)16 * LDS_ST * sizeof(__bf16);       //  8448 B
  const size_t smem_inner = (size_t)6 * 16 * LDS_ST * sizeof(__bf16);   // 50688 B

  for (int l = L - 1; l >= 0; --l) {
    int s = starts[l], sz = sizes[l];
    int blocks = (sz + 15) / 16;
    if (l == L - 1) {
      tree_level_kernel<true><<<blocks, 256, smem_leaf, stream>>>(
          tokens, emb, (const v16bf*)Wioux_p, (const v16bf*)Wiouh_p,
          (const v16bf*)Wfx_p, (const v16bf*)Wfh_p,
          bioux, biouh, bfx, bfh, h, cbuf, s, sz, N);
    } else {
      tree_level_kernel<false><<<blocks, 256, smem_inner, stream>>>(
          tokens, emb, (const v16bf*)Wioux_p, (const v16bf*)Wiouh_p,
          (const v16bf*)Wfx_p, (const v16bf*)Wfh_p,
          bioux, biouh, bfx, bfh, h, cbuf, s, sz, N);
    }
  }
}
